// IntensityNet_18468359373188
// MI455X (gfx1250) — matrix operand order repaired and checked by reading, not compiled, not run on hardware
//
#include <hip/hip_runtime.h>

// ---------------------------------------------------------------------------
// IntensityNet forward + JVP (grad w.r.t. target_time) for MI455X (gfx1250).
// Compute-bound GEMM chain -> f16 activations/weights, f32 accumulation via
// v_wmma_f32_16x16x32_f16. Layer-1 tangent is rank-1 (analytic); mid layers
// run Y and dY GEMMs fused so each weight fragment is used twice.
// global_prefetch of next K-step fragments at WGP scope (locality 3).
// ---------------------------------------------------------------------------

#define B_N   16384
#define T_N   32
#define HID_N 512
#define MLP_N 1024
#define EPSF  1e-10f

typedef __attribute__((ext_vector_type(16))) _Float16 v16h;
typedef __attribute__((ext_vector_type(8)))  _Float16 v8h;
typedef __attribute__((ext_vector_type(8)))  float    v8f;

__device__ __forceinline__ v16h join8(v8h lo, v8h hi) {
  return __builtin_shufflevector(lo, hi, 0,1,2,3,4,5,6,7,8,9,10,11,12,13,14,15);
}

// A-fragment (16x32 f16, row-major source, ld elements per row).
// ISA layout: lane<16 -> row=lane, K=0..7 then 16..23 ; lane>=16 -> K=8..15, 24..31.
__device__ __forceinline__ const _Float16* frag_a_ptr(const _Float16* __restrict__ base,
                                                      int ld, int row0, int k0, int lane) {
  return base + (size_t)(row0 + (lane & 15)) * ld + (k0 + ((lane >> 4) << 3));
}
__device__ __forceinline__ v16h load_frag_a(const _Float16* __restrict__ base,
                                            int ld, int row0, int k0, int lane) {
  const _Float16* p = frag_a_ptr(base, ld, row0, k0, lane);
  v8h lo = *(const v8h*)(p);
  v8h hi = *(const v8h*)(p + 16);
  return join8(lo, hi);
}

// B-fragment (32x16 f16). Source is W^T: N-major, K contiguous (ld = K).
// ISA layout: lane<16 -> col=lane, K=0..7 then 16..23 ; lane>=16 -> same col, K=8..15, 24..31.
__device__ __forceinline__ const _Float16* frag_b_ptr(const _Float16* __restrict__ baseT,
                                                      int ld, int col0, int k0, int lane) {
  return baseT + (size_t)(col0 + (lane & 15)) * ld + (k0 + ((lane >> 4) << 3));
}
__device__ __forceinline__ v16h load_frag_b(const _Float16* __restrict__ baseT,
                                            int ld, int col0, int k0, int lane) {
  const _Float16* p = frag_b_ptr(baseT, ld, col0, k0, lane);
  v8h lo = *(const v8h*)(p);
  v8h hi = *(const v8h*)(p + 16);
  return join8(lo, hi);
}

__device__ __forceinline__ v8f wmma16(v16h a, v16h b, v8f c) {
  return __builtin_amdgcn_wmma_f32_16x16x32_f16(false, a, false, b, (short)0, c, false, false);
}

__device__ __forceinline__ float softplusf(float x) {
  return fmaxf(x, 0.0f) + log1pf(expf(-fabsf(x)));
}
__device__ __forceinline__ float sigmoidf(float x) {
  return 1.0f / (1.0f + expf(-x));
}

// ---------------------------------------------------------------------------
// Prep kernels
// ---------------------------------------------------------------------------

// h_last extraction (f32 -> f16) + tfeat = t*w1 + b1
__global__ void k_extract(const float* __restrict__ hidden,
                          const float* __restrict__ tt,
                          const float* __restrict__ w1p,
                          const float* __restrict__ b1p,
                          _Float16* __restrict__ hlast,
                          float* __restrict__ tfeat) {
  int idx = blockIdx.x * 256 + threadIdx.x;          // B_N * HID_N threads
  int b = idx >> 9, h = idx & (HID_N - 1);
  hlast[idx] = (_Float16)hidden[(size_t)b * (T_N * HID_N) + (T_N - 1) * HID_N + h];
  if (h == 0) tfeat[b] = tt[b] * w1p[0] + b1p[0];
}

// Tiled transpose: src (K x N, f32 row-major) -> dst (N x K, f16 row-major)
__global__ void k_transpose(const float* __restrict__ src, _Float16* __restrict__ dst,
                            int K, int N) {
  __shared__ float tile[32][33];
  int k0 = blockIdx.x * 32, n0 = blockIdx.y * 32;
  int tx = threadIdx.x, ty = threadIdx.y;            // (32, 8)
#pragma unroll
  for (int j = 0; j < 4; ++j)
    tile[ty + j * 8][tx] = src[(size_t)(k0 + ty + j * 8) * N + n0 + tx];
  __syncthreads();
#pragma unroll
  for (int j = 0; j < 4; ++j)
    dst[(size_t)(n0 + ty + j * 8) * K + k0 + tx] = (_Float16)tile[tx][ty + j * 8];
}

__global__ void k_w2last(const float* __restrict__ w2, float* __restrict__ w2last) {
  int n = blockIdx.x * 256 + threadIdx.x;            // 1024 threads
  if (n < MLP_N) w2last[n] = w2[(size_t)HID_N * MLP_N + n];
}

// ---------------------------------------------------------------------------
// Layer 1: out = tanh(h_last @ W2a + tfeat*w2last + b2)
//          dout = (1 - out^2) * w1 * w2last          (rank-1 tangent)
// Wave tile 32x64 (MT=2 x NT=4), WG = 8 waves -> 128x128 block tile.
// ---------------------------------------------------------------------------
#define MT 2
#define NT 4

__global__ __launch_bounds__(256, 1)
void k_layer1(const _Float16* __restrict__ hlast,   // B_N x HID_N
              const float* __restrict__ tfeat,
              const _Float16* __restrict__ w2aT,    // MLP_N x HID_N (transposed)
              const float* __restrict__ w2last,
              const float* __restrict__ b2,
              const float* __restrict__ w1p,
              _Float16* __restrict__ outO,          // B_N x MLP_N
              _Float16* __restrict__ outD) {
  const int K = HID_N, N = MLP_N;
  int lane = threadIdx.x & 31, wid = threadIdx.x >> 5;
  int rowBase = blockIdx.y * 128 + (wid & 3) * 32;
  int colBase = blockIdx.x * 128 + (wid >> 2) * 64;

  v8f acc[MT][NT];
  const v8f zf = {0.f, 0.f, 0.f, 0.f, 0.f, 0.f, 0.f, 0.f};
#pragma unroll
  for (int mt = 0; mt < MT; ++mt)
#pragma unroll
    for (int nt = 0; nt < NT; ++nt) acc[mt][nt] = zf;

  for (int k0 = 0; k0 < K; k0 += 32) {
    v16h a[MT], b[NT];
#pragma unroll
    for (int mt = 0; mt < MT; ++mt) a[mt] = load_frag_a(hlast, K, rowBase + mt * 16, k0, lane);
#pragma unroll
    for (int nt = 0; nt < NT; ++nt) b[nt] = load_frag_b(w2aT, K, colBase + nt * 16, k0, lane);
    if (k0 + 32 < K) {  // WGP-scope prefetch of next K-step lines (counter-free)
#pragma unroll
      for (int mt = 0; mt < MT; ++mt)
        __builtin_prefetch(frag_a_ptr(hlast, K, rowBase + mt * 16, k0 + 32, lane), 0, 3);
#pragma unroll
      for (int nt = 0; nt < NT; ++nt)
        __builtin_prefetch(frag_b_ptr(w2aT, K, colBase + nt * 16, k0 + 32, lane), 0, 3);
    }
#pragma unroll
    for (int mt = 0; mt < MT; ++mt)
#pragma unroll
      for (int nt = 0; nt < NT; ++nt) acc[mt][nt] = wmma16(a[mt], b[nt], acc[mt][nt]);
  }

  float w1v = w1p[0];
  int n = lane & 15, rb = (lane >> 4) * 8;
#pragma unroll
  for (int nt = 0; nt < NT; ++nt) {
    int c = colBase + nt * 16 + n;
    float wl = w2last[c], bv = b2[c], dconst = w1v * wl;
#pragma unroll
    for (int mt = 0; mt < MT; ++mt) {
#pragma unroll
      for (int m = 0; m < 8; ++m) {
        int r = rowBase + mt * 16 + rb + m;
        float y = acc[mt][nt][m] + tfeat[r] * wl + bv;
        float o = tanhf(y);
        size_t idx = (size_t)r * N + c;
        outO[idx] = (_Float16)o;
        outD[idx] = (_Float16)((1.0f - o * o) * dconst);
      }
    }
  }
}

// ---------------------------------------------------------------------------
// Mid layer: Y = in @ W ; dY = din @ W  (shared B-fragments)
//   out = tanh(Y + b) ; dout = (1 - out^2) * dY
// ---------------------------------------------------------------------------
__global__ __launch_bounds__(256, 1)
void k_mid(const _Float16* __restrict__ inO, const _Float16* __restrict__ inD,
           const _Float16* __restrict__ WT,   // MLP_N x MLP_N, transposed
           const float* __restrict__ bias,
           _Float16* __restrict__ outO, _Float16* __restrict__ outD) {
  const int K = MLP_N, N = MLP_N;
  int lane = threadIdx.x & 31, wid = threadIdx.x >> 5;
  int rowBase = blockIdx.y * 128 + (wid & 3) * 32;
  int colBase = blockIdx.x * 128 + (wid >> 2) * 64;

  v8f accY[MT][NT], accD[MT][NT];
  const v8f zf = {0.f, 0.f, 0.f, 0.f, 0.f, 0.f, 0.f, 0.f};
#pragma unroll
  for (int mt = 0; mt < MT; ++mt)
#pragma unroll
    for (int nt = 0; nt < NT; ++nt) { accY[mt][nt] = zf; accD[mt][nt] = zf; }

  for (int k0 = 0; k0 < K; k0 += 32) {
    v16h a[MT], d[MT], b[NT];
#pragma unroll
    for (int mt = 0; mt < MT; ++mt) {
      a[mt] = load_frag_a(inO, K, rowBase + mt * 16, k0, lane);
      d[mt] = load_frag_a(inD, K, rowBase + mt * 16, k0, lane);
    }
#pragma unroll
    for (int nt = 0; nt < NT; ++nt) b[nt] = load_frag_b(WT, K, colBase + nt * 16, k0, lane);
    if (k0 + 32 < K) {  // WGP-scope prefetch of next K-step fragment lines
#pragma unroll
      for (int mt = 0; mt < MT; ++mt) {
        __builtin_prefetch(frag_a_ptr(inO, K, rowBase + mt * 16, k0 + 32, lane), 0, 3);
        __builtin_prefetch(frag_a_ptr(inD, K, rowBase + mt * 16, k0 + 32, lane), 0, 3);
      }
#pragma unroll
      for (int nt = 0; nt < NT; ++nt)
        __builtin_prefetch(frag_b_ptr(WT, K, colBase + nt * 16, k0 + 32, lane), 0, 3);
    }
#pragma unroll
    for (int mt = 0; mt < MT; ++mt)
#pragma unroll
      for (int nt = 0; nt < NT; ++nt) {
        accY[mt][nt] = wmma16(a[mt], b[nt], accY[mt][nt]);
        accD[mt][nt] = wmma16(d[mt], b[nt], accD[mt][nt]);
      }
  }

  int n = lane & 15, rb = (lane >> 4) * 8;
#pragma unroll
  for (int nt = 0; nt < NT; ++nt) {
    int c = colBase + nt * 16 + n;
    float bv = bias[c];
#pragma unroll
    for (int mt = 0; mt < MT; ++mt) {
#pragma unroll
      for (int m = 0; m < 8; ++m) {
        int r = rowBase + mt * 16 + rb + m;
        float o = tanhf(accY[mt][nt][m] + bv);
        size_t idx = (size_t)r * N + c;
        outO[idx] = (_Float16)o;
        outD[idx] = (_Float16)((1.0f - o * o) * accD[mt][nt][m]);
      }
    }
  }
}

// ---------------------------------------------------------------------------
// Final head: per-row dots with w3, softplus chain, lmbda, block partial sums.
// One wave per row; vectorized b128 loads; deterministic (no float atomics).
// ---------------------------------------------------------------------------
__global__ __launch_bounds__(256)
void k_final(const _Float16* __restrict__ out3, const _Float16* __restrict__ dout3,
             const float* __restrict__ w3, const float* __restrict__ b3,
             float* __restrict__ lmbda, float* __restrict__ partials) {
  int lane = threadIdx.x & 31, wid = threadIdx.x >> 5;
  int row = blockIdx.x * 8 + wid;
  const v8h* po8 = (const v8h*)(out3 + (size_t)row * MLP_N + lane * 32);
  const v8h* pd8 = (const v8h*)(dout3 + (size_t)row * MLP_N + lane * 32);
  const float* pw = w3 + lane * 32;
  float s = 0.f, t = 0.f;
#pragma unroll
  for (int c = 0; c < 4; ++c) {
    v8h ov = po8[c], dv = pd8[c];
#pragma unroll
    for (int j = 0; j < 8; ++j) {
      float w = pw[c * 8 + j];
      s += (float)ov[j] * w;
      t += (float)dv[j] * w;
    }
  }
#pragma unroll
  for (int off = 16; off > 0; off >>= 1) {
    s += __shfl_xor(s, off, 32);
    t += __shfl_xor(t, off, 32);
  }
  __shared__ float sil[8], sll[8];
  if (lane == 0) {
    float z   = s + b3[0];
    float sp1 = softplusf(z);
    float il  = softplusf(sp1);
    float lam = sigmoidf(sp1) * sigmoidf(z) * t * (1.0f / (float)B_N);
    lmbda[row] = lam;
    sil[wid] = il;
    sll[wid] = logf(lam + EPSF);
  }
  __syncthreads();
  if (threadIdx.x == 0) {
    float a = 0.f, b = 0.f;
#pragma unroll
    for (int i = 0; i < 8; ++i) { a += sil[i]; b += sll[i]; }
    partials[blockIdx.x * 2 + 0] = a;
    partials[blockIdx.x * 2 + 1] = b;
  }
}

__global__ __launch_bounds__(256)
void k_reduce(const float* __restrict__ partials, int nblk, float* __restrict__ out) {
  __shared__ float sA[256], sB[256];
  float a = 0.f, b = 0.f;
  for (int i = threadIdx.x; i < nblk; i += 256) {
    a += partials[2 * i + 0];
    b += partials[2 * i + 1];
  }
  sA[threadIdx.x] = a; sB[threadIdx.x] = b;
  __syncthreads();
  for (int s = 128; s > 0; s >>= 1) {
    if (threadIdx.x < (unsigned)s) {
      sA[threadIdx.x] += sA[threadIdx.x + s];
      sB[threadIdx.x] += sB[threadIdx.x + s];
    }
    __syncthreads();
  }
  if (threadIdx.x == 0) {
    float il_mean = sA[0] * (1.0f / (float)B_N);
    float ll_mean = sB[0] * (1.0f / (float)B_N);
    out[0] = il_mean - ll_mean;  // nll
    out[1] = ll_mean;            // log_lmbda_mean
    out[2] = il_mean;            // il_mean
  }
}

// ---------------------------------------------------------------------------
extern "C" void kernel_launch(void* const* d_in, const int* in_sizes, int n_in,
                              void* d_out, int out_size, void* d_ws, size_t ws_size,
                              hipStream_t stream) {
  const float* hidden = (const float*)d_in[0];
  const float* tt     = (const float*)d_in[1];
  const float* w1     = (const float*)d_in[2];
  const float* b1     = (const float*)d_in[3];
  const float* w2     = (const float*)d_in[4];
  const float* b2     = (const float*)d_in[5];
  const float* wsmid  = (const float*)d_in[6];
  const float* bsmid  = (const float*)d_in[7];
  const float* w3     = (const float*)d_in[8];
  const float* b3     = (const float*)d_in[9];
  float* out = (float*)d_out;

  char* ws = (char*)d_ws;
  size_t off = 0;
  auto take = [&](size_t bytes) -> void* {
    void* p = ws + off;
    off += (bytes + 255) & ~(size_t)255;
    return p;
  };
  _Float16* hlast  = (_Float16*)take((size_t)B_N * HID_N * 2);
  float*    tfeat  = (float*)   take((size_t)B_N * 4);
  _Float16* w2aT   = (_Float16*)take((size_t)MLP_N * HID_N * 2);
  float*    w2last = (float*)   take((size_t)MLP_N * 4);
  _Float16* wmidT  = (_Float16*)take((size_t)3 * MLP_N * MLP_N * 2);
  _Float16* outA   = (_Float16*)take((size_t)B_N * MLP_N * 2);
  _Float16* doutA  = (_Float16*)take((size_t)B_N * MLP_N * 2);
  _Float16* outB   = (_Float16*)take((size_t)B_N * MLP_N * 2);
  _Float16* doutB  = (_Float16*)take((size_t)B_N * MLP_N * 2);
  const int nFinalBlk = B_N / 8;                       // 2048
  float* partials = (float*)take((size_t)nFinalBlk * 2 * 4);

  // --- prep ---
  k_extract<<<(B_N * HID_N) / 256, 256, 0, stream>>>(hidden, tt, w1, b1, hlast, tfeat);
  k_transpose<<<dim3(HID_N / 32, MLP_N / 32), dim3(32, 8), 0, stream>>>(w2, w2aT, HID_N, MLP_N);
  k_w2last<<<MLP_N / 256, 256, 0, stream>>>(w2, w2last);
  for (int i = 0; i < 3; ++i)
    k_transpose<<<dim3(MLP_N / 32, MLP_N / 32), dim3(32, 8), 0, stream>>>(
        wsmid + (size_t)i * MLP_N * MLP_N, wmidT + (size_t)i * MLP_N * MLP_N, MLP_N, MLP_N);

  // --- MLP chain (forward + JVP fused) ---
  dim3 grid(MLP_N / 128, B_N / 128);  // (8, 128)
  k_layer1<<<grid, 256, 0, stream>>>(hlast, tfeat, w2aT, w2last, b2, w1, outA, doutA);
  k_mid<<<grid, 256, 0, stream>>>(outA, doutA, wmidT + 0 * MLP_N * MLP_N, bsmid + 0 * MLP_N, outB, doutB);
  k_mid<<<grid, 256, 0, stream>>>(outB, doutB, wmidT + 1 * MLP_N * MLP_N, bsmid + 1 * MLP_N, outA, doutA);
  k_mid<<<grid, 256, 0, stream>>>(outA, doutA, wmidT + 2 * MLP_N * MLP_N, bsmid + 2 * MLP_N, outB, doutB);

  // --- head + reductions ---
  k_final<<<nFinalBlk, 256, 0, stream>>>(outB, doutB, w3, b3, out + 3, partials);
  k_reduce<<<1, 256, 0, stream>>>(partials, nFinalBlk, out);
}
